// Decoder_74577812127757
// MI455X (gfx1250) — compile-verified, matrix operand order checked by reading, not run on hardware
//
#include <hip/hip_runtime.h>

#define B_N   65536
#define H_N   128
#define E_N   64
#define T_N   32
#define G_N   512      // 4*H
#define ROWS  16       // batch rows per workgroup
#define THREADS 512    // 16 waves (wave32)

typedef __attribute__((ext_vector_type(16))) _Float16 v16h;
typedef __attribute__((ext_vector_type(8)))  float    v8f;

__device__ __forceinline__ float fast_sigmoid(float x) {
    float e = __expf(-x);
    return __builtin_amdgcn_rcpf(1.0f + e);
}
__device__ __forceinline__ float fast_tanh(float x) {
    float e = __expf(2.0f * x);
    return 1.0f - 2.0f * __builtin_amdgcn_rcpf(e + 1.0f);
}

// ---------------------------------------------------------------------------
// Kernel 0: fold embedding into the LSTM input weights (rank-2 collapse).
//   M[0][j] = sum_e W_ih[j,e]*W_emb[e,0]
//   M[1][j] = sum_e W_ih[j,e]*W_emb[e,1]
//   M[2][j] = sum_e W_ih[j,e]*W_emb[e,2]
//   M[3][j] = b_ih[j] + b_hh[j] + sum_e W_ih[j,e]*b_emb[e]
// ---------------------------------------------------------------------------
__global__ void fold_weights(const float* __restrict__ W_ih,
                             const float* __restrict__ W_emb,
                             const float* __restrict__ b_emb,
                             const float* __restrict__ b_ih,
                             const float* __restrict__ b_hh,
                             float* __restrict__ Mv) {
    int j = threadIdx.x;   // 0..511
    const float* wr = W_ih + j * E_N;
    float a0 = 0.f, a1 = 0.f, a2 = 0.f, ab = 0.f;
    for (int e = 0; e < E_N; ++e) {
        float w = wr[e];
        a0 = fmaf(w, W_emb[e * 3 + 0], a0);
        a1 = fmaf(w, W_emb[e * 3 + 1], a1);
        a2 = fmaf(w, W_emb[e * 3 + 2], a2);
        ab = fmaf(w, b_emb[e], ab);
    }
    Mv[0 * G_N + j] = a0;
    Mv[1 * G_N + j] = a1;
    Mv[2 * G_N + j] = a2;
    Mv[3 * G_N + j] = b_ih[j] + b_hh[j] + ab;
}

// ---------------------------------------------------------------------------
// Fused decoder: WMMA prologue (base = h0 @ W_hh^T + affine) + 32-step
// wave-local recurrence.
// ---------------------------------------------------------------------------
__global__ __launch_bounds__(THREADS)
void decoder_fused(const float* __restrict__ last_pos_rel,
                   const float* __restrict__ h0,
                   const float* __restrict__ c0,
                   const float* __restrict__ speed,     // last_speed_pos_rel
                   const float* __restrict__ W_hh,      // [512,128]
                   const float* __restrict__ W_pos,     // [2,128]
                   const float* __restrict__ b_pos,     // [2]
                   const float* __restrict__ Mv,        // [4][512]
                   float* __restrict__ rels)            // [T,B,2]
{
    __shared__ float base_lds[ROWS * G_N];     // 32 KB
    __shared__ float rel_lds[T_N * ROWS * 2];  // 4 KB

    const int tid     = threadIdx.x;
    const int wave    = tid >> 5;
    const int lane    = tid & 31;
    const int rowBase = blockIdx.x * ROWS;

    // ---- Prologue: base[16x512] = h0_tile[16x128] @ W_hh^T via WMMA f16 ----
    // ISA 16-bit A/B layout: lane<16 holds halves K={k0..k0+7, k0+16..k0+23},
    // lane>=16 holds K={k0+8..k0+15, k0+24..k0+31}.
    const int mrow  = lane & 15;
    const int khalf = (lane >> 4) * 8;   // 0 or 8
    const float* M2v = Mv + 2 * G_N;
    const float* Kvv = Mv + 3 * G_N;

    for (int nt = 0; nt < 2; ++nt) {
        const int ntile = wave * 2 + nt;     // 16 waves x 2 = 32 N-tiles
        v8f acc = {};
        #pragma unroll
        for (int kc = 0; kc < 4; ++kc) {     // K = 128 in 4 chunks of 32
            const int k0 = kc * 32;
            const float* ar = h0 + (size_t)(rowBase + mrow) * H_N + k0;
            // B column n = W_hh row j (contiguous in memory)
            const int j     = ntile * 16 + mrow;
            const float* br = W_hh + (size_t)j * H_N + k0;
            v16h a, b;
            #pragma unroll
            for (int i = 0; i < 8; ++i) a[i]     = (_Float16)ar[khalf + i];
            #pragma unroll
            for (int i = 0; i < 8; ++i) a[8 + i] = (_Float16)ar[16 + khalf + i];
            #pragma unroll
            for (int i = 0; i < 8; ++i) b[i]     = (_Float16)br[khalf + i];
            #pragma unroll
            for (int i = 0; i < 8; ++i) b[8 + i] = (_Float16)br[16 + khalf + i];
            acc = __builtin_amdgcn_wmma_f32_16x16x32_f16(
                      false, a, false, b, (short)0, acc, false, false);
        }
        // Scatter D (16x16 f32) to LDS, folding speed*M2 + K terms.
        const int n  = ntile * 16 + (lane & 15);
        const float m2 = M2v[n], kv = Kvv[n];
        #pragma unroll
        for (int r = 0; r < 8; ++r) {
            int m = r + ((lane >> 4) * 8);   // lanes 0-15: M=r ; 16-31: M=8+r
            float sp = speed[rowBase + m];
            base_lds[m * G_N + n] = acc[r] + sp * m2 + kv;
        }
    }
    __syncthreads();

    // ---- Recurrence: wave w <-> row w, lane l <-> hidden 4l..4l+3 ----
    const int row  = wave;
    const int h    = lane * 4;
    const int gRow = rowBase + row;
    const float* M0v = Mv;
    const float* M1v = Mv + G_N;

    float bi[4], bf[4], bg[4], bo[4];
    float m0i[4], m0f[4], m0g[4], m0o[4];
    float m1i[4], m1f[4], m1g[4], m1o[4];
    float cc[4], wp0[4], wp1[4];
    #pragma unroll
    for (int k = 0; k < 4; ++k) {
        int ji = h + k, jf = 128 + h + k, jg = 256 + h + k, jo = 384 + h + k;
        bi[k] = base_lds[row * G_N + ji]; bf[k] = base_lds[row * G_N + jf];
        bg[k] = base_lds[row * G_N + jg]; bo[k] = base_lds[row * G_N + jo];
        m0i[k] = M0v[ji]; m0f[k] = M0v[jf]; m0g[k] = M0v[jg]; m0o[k] = M0v[jo];
        m1i[k] = M1v[ji]; m1f[k] = M1v[jf]; m1g[k] = M1v[jg]; m1o[k] = M1v[jo];
        cc[k]  = c0[(size_t)gRow * H_N + h + k];
        wp0[k] = W_pos[h + k];
        wp1[k] = W_pos[H_N + h + k];
    }
    float rel0 = last_pos_rel[gRow * 2 + 0];
    float rel1 = last_pos_rel[gRow * 2 + 1];
    const float bp0 = b_pos[0], bp1 = b_pos[1];

    for (int t = 0; t < T_N; ++t) {
        float s0 = 0.f, s1 = 0.f;
        #pragma unroll
        for (int k = 0; k < 4; ++k) {
            float gi = fmaf(rel1, m1i[k], fmaf(rel0, m0i[k], bi[k]));
            float gf = fmaf(rel1, m1f[k], fmaf(rel0, m0f[k], bf[k]));
            float gg = fmaf(rel1, m1g[k], fmaf(rel0, m0g[k], bg[k]));
            float go = fmaf(rel1, m1o[k], fmaf(rel0, m0o[k], bo[k]));
            float cn = fast_sigmoid(gf) * cc[k] + fast_sigmoid(gi) * fast_tanh(gg);
            float hn = fast_sigmoid(go) * fast_tanh(cn);
            s0 = fmaf(hn, wp0[k], s0);
            s1 = fmaf(hn, wp1[k], s1);
        }
        // full-wave butterfly reduction over 32 lanes (all lanes get result)
        #pragma unroll
        for (int off = 16; off > 0; off >>= 1) {
            s0 += __shfl_xor(s0, off, 32);
            s1 += __shfl_xor(s1, off, 32);
        }
        rel0 = s0 + bp0;
        rel1 = s1 + bp1;
        if (lane < 2) rel_lds[(t * ROWS + row) * 2 + lane] = lane ? rel1 : rel0;
    }
    __syncthreads();

    // coalesced write: per t, the block's 16 rows are 32 contiguous floats
    for (int idx = tid; idx < T_N * ROWS * 2; idx += THREADS) {
        int t = idx >> 5;        // ROWS*2 == 32
        int r = idx & 31;
        rels[(size_t)t * (B_N * 2) + rowBase * 2 + r] = rel_lds[idx];
    }
}

extern "C" void kernel_launch(void* const* d_in, const int* in_sizes, int n_in,
                              void* d_out, int out_size, void* d_ws, size_t ws_size,
                              hipStream_t stream) {
    (void)in_sizes; (void)n_in; (void)out_size; (void)ws_size;
    // inputs in setup_inputs() order
    const float* last_pos_rel = (const float*)d_in[1];
    const float* h0           = (const float*)d_in[2];
    const float* c0           = (const float*)d_in[3];
    const float* speed        = (const float*)d_in[4];  // last_speed_pos_rel
    const float* W_ih         = (const float*)d_in[7];
    const float* W_hh         = (const float*)d_in[8];
    const float* b_ih         = (const float*)d_in[9];
    const float* b_hh         = (const float*)d_in[10];
    const float* W_emb        = (const float*)d_in[11];
    const float* b_emb        = (const float*)d_in[12];
    const float* W_pos        = (const float*)d_in[13];
    const float* b_pos        = (const float*)d_in[14];
    // d_in[0] last_pos, d_in[5] last_speed_abs_pos, d_in[6] seq_len: unused
    // (last_pos provably never affects rels; seq_len is the fixed T=32)

    float* Mv  = (float*)d_ws;     // 4*512 f32 = 8 KB scratch
    float* out = (float*)d_out;

    fold_weights<<<1, G_N, 0, stream>>>(W_ih, W_emb, b_emb, b_ih, b_hh, Mv);
    decoder_fused<<<B_N / ROWS, THREADS, 0, stream>>>(
        last_pos_rel, h0, c0, speed, W_hh, W_pos, b_pos, Mv, out);
    // second output: state_tuple[0] == h0[0:1], appended after rels
    hipMemcpyAsync(out + (size_t)T_N * B_N * 2, h0,
                   (size_t)B_N * H_N * sizeof(float),
                   hipMemcpyDeviceToDevice, stream);
}
